// Component_prototypes_35734127902888
// MI455X (gfx1250) — matrix operand order repaired and checked by reading, not compiled, not run on hardware
//
#include <hip/hip_runtime.h>
#include <hip/hip_bf16.h>
#include <math.h>

// ---------------------------------------------------------------------------
// Problem constants (from the reference)
// ---------------------------------------------------------------------------
#define BB 1024          // batch
#define DD 128           // feature dim
#define LL 34            // N_AFE_A + N_AFE_R*S = 2 + 32
#define AA 4096          // atoms
#define MROWS (BB * LL)  // 34816 flattened (b,l) rows; 34816/32 = 1088 exact

typedef __attribute__((ext_vector_type(16))) _Float16 v16h;
typedef __attribute__((ext_vector_type(8)))  float    v8f;

// ---------------------------------------------------------------------------
// Kernel 0: normalize atoms -> f16 atoms_n (for WMMA B operand) + inv-norm f32
// ---------------------------------------------------------------------------
__global__ __launch_bounds__(128) void atoms_norm_kernel(
    const float* __restrict__ atoms,    // (A, D)
    float* __restrict__ atom_inv,       // (A,)
    _Float16* __restrict__ atoms_h)     // (A, D) f16 normalized
{
    const int a = blockIdx.x;
    const int t = threadIdx.x;                 // 0..127 == d
    const float x = atoms[(size_t)a * DD + t];
    float ss = x * x;
    #pragma unroll
    for (int off = 16; off; off >>= 1) ss += __shfl_xor(ss, off, 32);
    __shared__ float red[4];
    if ((t & 31) == 0) red[t >> 5] = ss;
    __syncthreads();
    const float tot   = red[0] + red[1] + red[2] + red[3];
    const float scale = 1.0f / fmaxf(sqrtf(tot), 1e-12f);   // matches reference eps
    atoms_h[(size_t)a * DD + t] = (_Float16)(x * scale);
    if (t == 0) atom_inv[a] = scale;
}

// ---------------------------------------------------------------------------
// Kernel 1: score GEMM (E . atoms_n^T) with fused per-row argmax.
// One wave per 32-row M block (two 16x128 A-fragment sets resident in VGPRs).
// N = 4096 atoms = 256 tiles of 16; B fragments double-buffered so tile n+1's
// global loads overlap tile n's two independent WMMA chains. K = 128 = 4
// k-steps of 32. Prefetch runs 8 tiles ahead of the double buffer.
// ---------------------------------------------------------------------------
__global__ __launch_bounds__(32) void score_argmax_kernel(
    const float* __restrict__ emb,        // (MROWS, D) f32
    const _Float16* __restrict__ atoms_h, // (A, D) f16 normalized
    int* __restrict__ idx)                // (MROWS,)
{
    const int lane = threadIdx.x;       // 0..31
    const int r    = lane & 15;
    const int hs   = lane >> 4;         // lane-half select
    const int m0   = blockIdx.x * 32;   // base row (always in range)

    // --- A fragments: 16x32 f16 per k-step, resident across the whole N loop.
    // Layout chosen here (same for A and B): per lane, elems 0..15 -> K = kbase..kbase+15,
    //             kbase = 16*hs (contiguous, matches B).
    const int kbase = hs * 16;
    v16h a_frag[2][4];
    #pragma unroll
    for (int mt = 0; mt < 2; ++mt) {
        const float* erow = emb + (size_t)(m0 + mt * 16 + r) * DD;
        #pragma unroll
        for (int ks = 0; ks < 4; ++ks) {
            const float* p = erow + ks * 32 + kbase;
            v16h a;
            #pragma unroll
            for (int i = 0; i < 8; ++i) a[i]     = (_Float16)p[i];
            #pragma unroll
            for (int i = 0; i < 8; ++i) a[8 + i] = (_Float16)p[8 + i];
            a_frag[mt][ks] = a;
        }
    }

    // Running (max, argmax): lane holds rows M = mt*16 + hs*8 + vg, col n0 + r.
    float best[2][8];
    int   bi[2][8];
    #pragma unroll
    for (int mt = 0; mt < 2; ++mt)
        #pragma unroll
        for (int vg = 0; vg < 8; ++vg) { best[mt][vg] = -3.402823466e38f; bi[mt][vg] = 0; }

    // B fragment loader: lane n = r holds K = hs*16 + [0..15] contiguous halves.
    auto loadB = [&](int nt, v16h (&bf)[4]) {
        const _Float16* arow = atoms_h + (size_t)(nt * 16 + r) * DD + hs * 16;
        #pragma unroll
        for (int ks = 0; ks < 4; ++ks) bf[ks] = *(const v16h*)(arow + ks * 32);
    };

    // Two independent C chains per B tile (interleaved for XDL pipelining),
    // fused max/argmax epilogue (co-executes with the next tile's WMMAs).
    auto compute = [&](int nt, const v16h (&bf)[4]) {
        v8f c0 = {}, c1 = {};
        #pragma unroll
        for (int ks = 0; ks < 4; ++ks) {
            c0 = __builtin_amdgcn_wmma_f32_16x16x32_f16(
                     false, a_frag[0][ks], false, bf[ks], (short)0, c0, false, false);
            c1 = __builtin_amdgcn_wmma_f32_16x16x32_f16(
                     false, a_frag[1][ks], false, bf[ks], (short)0, c1, false, false);
        }
        const int col = nt * 16 + r;
        #pragma unroll
        for (int vg = 0; vg < 8; ++vg) {
            if (c0[vg] > best[0][vg]) { best[0][vg] = c0[vg]; bi[0][vg] = col; }
            if (c1[vg] > best[1][vg]) { best[1][vg] = c1[vg]; bi[1][vg] = col; }
        }
    };

    v16h b0[4], b1[4];
    loadB(0, b0);
    for (int nt = 0; nt < AA / 16; nt += 2) {
        loadB(nt + 1, b1);
        // Keep the atom stream warm ~8 tiles ahead (global_prefetch_b8).
        __builtin_prefetch(atoms_h + (size_t)((((nt + 8) & 255) * 16) + r) * DD, 0, 1);
        compute(nt, b0);
        loadB((nt + 2) & 255, b0);   // wraps harmlessly in-bounds on last iter
        compute(nt + 1, b1);
    }

    // Reduce across the 16 lanes of each half (xor masks stay within half);
    // lowest index wins ties, matching argmax semantics.
    #pragma unroll
    for (int off = 8; off; off >>= 1) {
        #pragma unroll
        for (int mt = 0; mt < 2; ++mt) {
            #pragma unroll
            for (int vg = 0; vg < 8; ++vg) {
                float ov = __shfl_xor(best[mt][vg], off, 32);
                int   oi = __shfl_xor(bi[mt][vg],   off, 32);
                if (ov > best[mt][vg] || (ov == best[mt][vg] && oi < bi[mt][vg])) {
                    best[mt][vg] = ov; bi[mt][vg] = oi;
                }
            }
        }
    }
    if (r == 0) {
        #pragma unroll
        for (int mt = 0; mt < 2; ++mt)
            #pragma unroll
            for (int vg = 0; vg < 8; ++vg)
                idx[m0 + mt * 16 + hs * 8 + vg] = bi[mt][vg];
    }
}

// ---------------------------------------------------------------------------
// Kernel 2: per-b epilogue. assoc[l] = atoms[idx]*inv_norm (== atoms_n, f32),
// att = assoc_flat @ W_att + b_att, obj = normalize(sum_l att[l]*assoc[l]).
// ---------------------------------------------------------------------------
__global__ __launch_bounds__(128) void assemble_kernel(
    const float* __restrict__ atoms,     // (A, D)
    const float* __restrict__ atom_inv,  // (A,)
    const int*   __restrict__ idx,       // (B, L)
    const float* __restrict__ W_att,     // (L*D, L) row-major
    const float* __restrict__ b_att,     // (L,)
    float* __restrict__ out)             // (B, D)
{
    __shared__ float s_assoc[LL * DD];   // 17.4 KB of the 320 KB WGP LDS
    __shared__ float s_att[LL];
    __shared__ float s_red[4];

    const int b = blockIdx.x;
    const int t = threadIdx.x;           // 0..127 == d

    // Gather normalized assoc rows into LDS.
    #pragma unroll 2
    for (int l = 0; l < LL; ++l) {
        const int a = idx[b * LL + l];
        s_assoc[l * DD + t] = atoms[(size_t)a * DD + t] * atom_inv[a];
    }
    if (t < LL) s_att[t] = b_att[t];
    __syncthreads();

    // att[j] = b_att[j] + sum_i assoc_flat[i] * W_att[i*L + j]
    float p[LL];
    #pragma unroll
    for (int j = 0; j < LL; ++j) p[j] = 0.0f;
    for (int i = t; i < LL * DD; i += 128) {
        const float x = s_assoc[i];
        const float* w = W_att + (size_t)i * LL;
        #pragma unroll
        for (int j = 0; j < LL; ++j) p[j] += x * w[j];
    }
    // Wave-reduce partials, then one ds_add_f32 per wave per output.
    #pragma unroll
    for (int j = 0; j < LL; ++j) {
        float v = p[j];
        #pragma unroll
        for (int off = 16; off; off >>= 1) v += __shfl_xor(v, off, 32);
        if ((t & 31) == 0) atomicAdd(&s_att[j], v);
    }
    __syncthreads();

    // obj[d] = sum_l att[l] * assoc[l][d], then L2-normalize over d.
    float acc = 0.0f;
    #pragma unroll
    for (int l = 0; l < LL; ++l) acc += s_att[l] * s_assoc[l * DD + t];

    float ss = acc * acc;
    #pragma unroll
    for (int off = 16; off; off >>= 1) ss += __shfl_xor(ss, off, 32);
    if ((t & 31) == 0) s_red[t >> 5] = ss;
    __syncthreads();
    const float tot   = s_red[0] + s_red[1] + s_red[2] + s_red[3];
    const float scale = 1.0f / fmaxf(sqrtf(tot), 1e-12f);
    out[(size_t)b * DD + t] = acc * scale;
}

// ---------------------------------------------------------------------------
// Launch
// ---------------------------------------------------------------------------
extern "C" void kernel_launch(void* const* d_in, const int* in_sizes, int n_in,
                              void* d_out, int out_size, void* d_ws, size_t ws_size,
                              hipStream_t stream) {
    const float* emb   = (const float*)d_in[0];   // (B, L, D)
    const float* atoms = (const float*)d_in[1];   // (A, D)
    const float* W_att = (const float*)d_in[2];   // (L*D, L)
    const float* b_att = (const float*)d_in[3];   // (L,)
    float* out = (float*)d_out;                   // (B, D)

    // Workspace layout (~1.2 MB total):
    float*    atom_inv = (float*)d_ws;                          // A f32
    _Float16* atoms_h  = (_Float16*)(atom_inv + AA);            // A*D f16
    int*      idx      = (int*)((char*)d_ws + AA * sizeof(float)
                                          + (size_t)AA * DD * sizeof(_Float16));

    atoms_norm_kernel<<<AA, 128, 0, stream>>>(atoms, atom_inv, atoms_h);
    score_argmax_kernel<<<MROWS / 32, 32, 0, stream>>>(emb, atoms_h, idx);
    assemble_kernel<<<BB, 128, 0, stream>>>(atoms, atom_inv, idx, W_att, b_att, out);
}
